// TimeAttention_50251117363232
// MI455X (gfx1250) — matrix operand order repaired and checked by reading, not compiled, not run on hardware
//
#include <hip/hip_runtime.h>
#include <math.h>

// ---------------- Problem constants (fixed shapes from setup_inputs) -------
#define NB   8
#define CC   256
#define HW_  16384
#define TK_  128

typedef __attribute__((ext_vector_type(16))) __bf16 v16bf;
typedef __attribute__((ext_vector_type(8)))  float  v8f;
typedef __attribute__((ext_vector_type(4)))  int    v4i;

#if defined(__gfx1250__) && __has_builtin(__builtin_amdgcn_global_load_async_to_lds_b128) && \
    __has_builtin(__builtin_amdgcn_s_wait_asynccnt)
#define USE_ASYNC_LDS 1
typedef __attribute__((address_space(1))) v4i GV4i;   // global int4
typedef __attribute__((address_space(3))) v4i LV4i;   // LDS int4
#else
#define USE_ASYNC_LDS 0
#endif

union Frag { v16bf v; uint4 q[2]; };

static __device__ __forceinline__ unsigned short f2bf(float f) {
  unsigned u = __builtin_bit_cast(unsigned, f);
  unsigned r = u + 0x7FFFu + ((u >> 16) & 1u);   // round-to-nearest-even
  return (unsigned short)(r >> 16);
}
static __device__ __forceinline__ float bf2f(unsigned short h) {
  return __builtin_bit_cast(float, (unsigned)h << 16);
}
static __device__ __forceinline__ v8f zero8() {
  v8f z; for (int i = 0; i < 8; ++i) z[i] = 0.0f; return z;
}

// ---------------- 1) W_conv fp32 -> bf16 -----------------------------------
__global__ void k_cvt_wconv(const float* __restrict__ w, unsigned short* __restrict__ o) {
  int i = blockIdx.x * 256 + threadIdx.x;       // 65536 elements
  o[i] = f2bf(w[i]);
}

// ---------------- 2) fc on object features + row norms (nb) ----------------
// pre_adj[row, d] = sum_c pre[row,c] * W_fc[d,c] + b_fc[d];  row = n*TK+tk
__global__ void k_prep(const float* __restrict__ pre, const float* __restrict__ W_fc,
                       const float* __restrict__ b_fc,
                       unsigned short* __restrict__ pre_bf, float* __restrict__ nb) {
  __shared__ float rowv[CC];
  __shared__ float red[256];
  int row = blockIdx.x;                 // 0..1023
  int d = threadIdx.x;                  // 0..255
  rowv[d] = pre[(size_t)row * CC + d];
  __syncthreads();
  float acc = b_fc[d];
  const float* w = W_fc + (size_t)d * CC;
  for (int c = 0; c < CC; ++c) acc = fmaf(rowv[c], w[c], acc);
  pre_bf[(size_t)row * CC + d] = f2bf(acc);
  red[d] = acc * acc;
  __syncthreads();
  for (int s = 128; s > 0; s >>= 1) { if (d < s) red[d] += red[d + s]; __syncthreads(); }
  if (d == 0) nb[row] = sqrtf(red[0]);
}

// ---------------- 3) 1x1 conv GEMM via bf16 WMMA ---------------------------
// ftadj_bf16[n, d, p] = bf16( sum_c W_conv[d,c]*ft[n,c,p] + b_conv[d] )
// Block: 256 thr = 8 waves; block tile 64(M=d) x 128(N=p); wave tile 16x64.
__global__ void k_conv(const float* __restrict__ ft, const unsigned short* __restrict__ wconv,
                       const float* __restrict__ bconv, unsigned short* __restrict__ ftadj) {
  __shared__ __align__(16) unsigned short ldsB[128 * 56];  // (p, c) tile, stride 56
  int bid = blockIdx.x;
  int n   = bid >> 9;                           // 512 blocks per n
  int rem = bid & 511;
  int mt  = rem >> 7;                           // 0..3  (M tile of 64)
  int nt  = rem & 127;                          // 0..127 (N tile of 128)
  int p0  = nt * 128;
  int t = threadIdx.x, wave = t >> 5, lane = t & 31;
  int wm = wave & 3, wn = wave >> 2;            // 4 M-strips x 2 N-halves
  int l15 = lane & 15, hi = lane >> 4;

  v8f acc[4]; for (int q = 0; q < 4; ++q) acc[q] = zero8();
  const float* ftn = ft + (size_t)n * CC * HW_;

  for (int kk = 0; kk < CC; kk += 32) {
    // stage B: 32 rows (c) x 128 cols (p) fp32 -> bf16 LDS transposed (p,c)
    for (int v = 0; v < 4; ++v) {
      int e4 = t + v * 256;                     // 0..1023 float4 slots
      int c  = e4 >> 5;                         // 0..31
      int p  = (e4 & 31) * 4;                   // 0..124
      const float* src = ftn + (size_t)(kk + c) * HW_ + p0 + p;
      float4 f = *(const float4*)src;
      if (kk + 32 < CC)
        __builtin_prefetch(src + 32 * HW_, 0, 3);   // next k-tile of this stream
      ldsB[(p + 0) * 56 + c] = f2bf(f.x);
      ldsB[(p + 1) * 56 + c] = f2bf(f.y);
      ldsB[(p + 2) * 56 + c] = f2bf(f.z);
      ldsB[(p + 3) * 56 + c] = f2bf(f.w);
    }
    __syncthreads();
    // A fragment: 16x32 from W_conv (row-major d,c)
    Frag a;
    {
      int d  = mt * 64 + wm * 16 + l15;
      int kb = hi ? 8 : 0;
      const unsigned short* ap = wconv + (size_t)d * CC + kk + kb;
      a.q[0] = *(const uint4*)(ap);
      a.q[1] = *(const uint4*)(ap + 16);
    }
    for (int q = 0; q < 4; ++q) {
      int p  = wn * 64 + q * 16 + l15;
      int kb = hi ? 8 : 0;
      const unsigned short* bp = &ldsB[p * 56 + kb];
      Frag b;
      b.q[0] = *(const uint4*)(bp);
      b.q[1] = *(const uint4*)(bp + 16);
      acc[q] = __builtin_amdgcn_wmma_f32_16x16x32_bf16(false, a.v, false, b.v,
                                                       (short)0, acc[q], false, false);
    }
    __syncthreads();
  }
  unsigned short* outn = ftadj + (size_t)n * CC * HW_;
  for (int q = 0; q < 4; ++q) {
    int p = p0 + wn * 64 + q * 16 + l15;
    for (int r = 0; r < 8; ++r) {
      int d = mt * 64 + wm * 16 + r + hi * 8;
      outn[(size_t)d * HW_ + p] = f2bf(acc[q][r] + bconv[d]);
    }
  }
}

// ---------------- 4) na[row] = ||ft_rows[row,:]|| (bf16 rows of 256) -------
__global__ void k_na(const unsigned short* __restrict__ ftadj, float* __restrict__ na) {
  int wave = threadIdx.x >> 5, lane = threadIdx.x & 31;
  int row = blockIdx.x * 8 + wave;              // 0..131071
  uint4 q = *(const uint4*)(ftadj + (size_t)row * CC + lane * 8);
  const unsigned short* u = (const unsigned short*)&q;
  float s = 0.f;
  for (int e = 0; e < 8; ++e) { float v = bf2f(u[e]); s = fmaf(v, v, s); }
  for (int m = 16; m > 0; m >>= 1) s += __shfl_xor(s, m, 32);
  if (lane == 0) na[row] = sqrtf(s);
}

// ---------------- 5) dots GEMM + cosine scale via bf16 WMMA ----------------
// sim[n, j, i] = (pre_adj[n,j,:] . ft_rows[n,i,:]) / max(nb*na, eps)
// Block covers full M=128 (one 16-row strip per wave) x 64 i-columns.
// B tile (64 x 32 bf16) staged once per block in LDS via async global->LDS
// copy (ASYNCcnt path) so all 8 waves share it instead of 8x-redundant
// global fragment loads.
__global__ void k_dots(const unsigned short* __restrict__ pre_bf,
                       const unsigned short* __restrict__ ftadj,
                       const float* __restrict__ nb, const float* __restrict__ na,
                       float* __restrict__ sim) {
  __shared__ __align__(16) unsigned short ldsB[64 * 40];   // (i, c) tile, stride 40
  int bid = blockIdx.x;
  int n  = bid >> 8;                            // 256 blocks per n
  int i0 = (bid & 255) * 64;
  int t = threadIdx.x, wave = t >> 5, lane = t & 31;
  int l15 = lane & 15, hi = lane >> 4;
  int j0 = wave * 16;

  v8f acc[4]; for (int q = 0; q < 4; ++q) acc[q] = zero8();
  const unsigned short* An = pre_bf + (size_t)n * TK_ * CC;
  const unsigned short* Bn = ftadj  + (size_t)n * CC * HW_;   // flat == (HW, C) rows

  int si = t >> 2;                              // staging row 0..63
  int sc = (t & 3) * 8;                         // ushort offset 0/8/16/24

  for (int kk = 0; kk < CC; kk += 32) {
    // stage B tile: 64 rows (i) x 32 (c) bf16, straight byte copy
    {
      const unsigned short* gsrc = Bn + (size_t)(i0 + si) * CC + kk + sc;
      unsigned short* ldst = &ldsB[si * 40 + sc];
#if USE_ASYNC_LDS
      __builtin_amdgcn_global_load_async_to_lds_b128((GV4i*)gsrc, (LV4i*)ldst, 0, 0);
#else
      *(uint4*)ldst = *(const uint4*)gsrc;
#endif
    }
    // A fragment: 16x32 from pre_adj (row-major j,c), tiny & L0-resident
    Frag a;
    {
      int kb = hi ? 8 : 0;
      const unsigned short* ap = An + (size_t)(j0 + l15) * CC + kk + kb;
      a.q[0] = *(const uint4*)(ap);
      a.q[1] = *(const uint4*)(ap + 16);
    }
#if USE_ASYNC_LDS
    __builtin_amdgcn_s_wait_asynccnt(0);
#endif
    __syncthreads();
    for (int q = 0; q < 4; ++q) {
      int kb = hi ? 8 : 0;
      const unsigned short* bp = &ldsB[(q * 16 + l15) * 40 + kb];
      Frag b;
      b.q[0] = *(const uint4*)(bp);
      b.q[1] = *(const uint4*)(bp + 16);
      acc[q] = __builtin_amdgcn_wmma_f32_16x16x32_bf16(false, a.v, false, b.v,
                                                       (short)0, acc[q], false, false);
    }
    __syncthreads();
  }
  float* simn = sim + (size_t)n * TK_ * HW_;
  for (int q = 0; q < 4; ++q) {
    int i = i0 + q * 16 + l15;
    float nav = na[(size_t)n * HW_ + i];
    for (int r = 0; r < 8; ++r) {
      int j = j0 + r + hi * 8;
      float den = fmaxf(nb[n * TK_ + j] * nav, 1e-8f);
      simn[(size_t)j * HW_ + i] = acc[q][r] / den;
    }
  }
}

// ---------------- 6) softmax stats per (n,j): max & sum(exp) over HW -------
__global__ void k_softmax_stats(const float* __restrict__ sim,
                                float* __restrict__ Mst, float* __restrict__ Sst) {
  __shared__ float red[256];
  int row = blockIdx.x;                         // n*TK+j, 0..1023
  const float* p = sim + (size_t)row * HW_;
  int t = threadIdx.x;
  float m = -1e30f;
  for (int k = t; k < HW_; k += 256) m = fmaxf(m, p[k]);
  red[t] = m; __syncthreads();
  for (int s = 128; s > 0; s >>= 1) { if (t < s) red[t] = fmaxf(red[t], red[t + s]); __syncthreads(); }
  float M = red[0]; __syncthreads();
  float sum = 0.f;
  for (int k = t; k < HW_; k += 256) sum += __expf(p[k] - M);
  red[t] = sum; __syncthreads();
  for (int s = 128; s > 0; s >>= 1) { if (t < s) red[t] += red[t + s]; __syncthreads(); }
  if (t == 0) { Mst[row] = M; Sst[row] = red[0]; }
}

// ---------------- 7) gate[n,i] = sum_j exp(sim - M[j]) / S[j] --------------
__global__ void k_gate(const float* __restrict__ sim, const float* __restrict__ Mst,
                       const float* __restrict__ Sst, float* __restrict__ gate) {
  int tid = blockIdx.x * 256 + threadIdx.x;     // 0..131071
  int n = tid >> 14;
  int i = tid & (HW_ - 1);
  const float* base = sim + (size_t)n * TK_ * HW_ + i;
  float acc = 0.f;
  for (int j = 0; j < TK_; ++j) {
    int row = n * TK_ + j;
    if (j + 8 < TK_)
      __builtin_prefetch(base + (size_t)(j + 8) * HW_, 0, 3);  // 64KB-stride column walk
    acc += __expf(base[(size_t)j * HW_] - Mst[row]) * __frcp_rn(Sst[row]);
  }
  gate[tid] = acc;
}

// ---------------- 8) out[n,c,p] = gate[n,p] * ft[n,c,p] --------------------
__global__ void k_apply(const float* __restrict__ ft, const float* __restrict__ gate,
                        float* __restrict__ out) {
  size_t tid = (size_t)blockIdx.x * 256 + threadIdx.x;  // float4 index
  size_t flat = tid * 4;
  int n = (int)(flat >> 22);                    // C*HW = 2^22
  int p = (int)(flat & (HW_ - 1));
  float4 f = ((const float4*)ft)[tid];
  float4 g = *(const float4*)(gate + ((size_t)n << 14) + p);
  float4 o;
  o.x = f.x * g.x; o.y = f.y * g.y; o.z = f.z * g.z; o.w = f.w * g.w;
  ((float4*)out)[tid] = o;
}

// ---------------------------------------------------------------------------
extern "C" void kernel_launch(void* const* d_in, const int* in_sizes, int n_in,
                              void* d_out, int out_size, void* d_ws, size_t ws_size,
                              hipStream_t stream) {
  const float* pre    = (const float*)d_in[0];  // (8,4,32,256)
  const float* ft     = (const float*)d_in[1];  // (8,256,128,128)
  const float* W_fc   = (const float*)d_in[2];  // (256,256)
  const float* b_fc   = (const float*)d_in[3];  // (256,)
  const float* W_conv = (const float*)d_in[4];  // (256,256)
  const float* b_conv = (const float*)d_in[5];  // (256,)
  float* out = (float*)d_out;

  char* ws = (char*)d_ws;
  unsigned short* wconv_bf = (unsigned short*)(ws);                 // 131072 B
  unsigned short* pre_bf   = (unsigned short*)(ws + 131072);        // 524288 B
  float* nb   = (float*)(ws + 655360);                              // 4096 B
  float* na   = (float*)(ws + 659456);                              // 524288 B
  unsigned short* ftadj = (unsigned short*)(ws + 1183744);          // 64 MiB
  float* sim  = (float*)(ws + 68292608);                            // 64 MiB
  float* Mst  = (float*)(ws + 135401472);                           // 4096 B
  float* Sst  = (float*)(ws + 135405568);                           // 4096 B
  float* gate = (float*)(ws + 135409664);                           // 524288 B

  k_cvt_wconv   <<<256,   256, 0, stream>>>(W_conv, wconv_bf);
  k_prep        <<<1024,  256, 0, stream>>>(pre, W_fc, b_fc, pre_bf, nb);
  k_conv        <<<4096,  256, 0, stream>>>(ft, wconv_bf, b_conv, ftadj);
  k_na          <<<16384, 256, 0, stream>>>(ftadj, na);
  k_dots        <<<2048,  256, 0, stream>>>(pre_bf, ftadj, nb, na, sim);
  k_softmax_stats<<<1024, 256, 0, stream>>>(sim, Mst, Sst);
  k_gate        <<<512,   256, 0, stream>>>(sim, Mst, Sst, gate);
  k_apply       <<<32768, 256, 0, stream>>>(ft, gate, out);
}
